// InfoNCELoss_42013370090174
// MI455X (gfx1250) — matrix operand order repaired and checked by reading, not compiled, not run on hardware
//
#include <hip/hip_runtime.h>
#include <math.h>

#define B_N 8192
#define D_K 512
#define NK  (D_K / 32)            // 16 k-steps
#define TEMP_INV (1.0f / 0.07f)
#define EPS_F 1e-8f

typedef __attribute__((ext_vector_type(16))) __bf16 v16bf;
typedef __attribute__((ext_vector_type(8)))  __bf16 v8bf;
typedef __attribute__((ext_vector_type(8)))  float  v8f;

__device__ __forceinline__ unsigned short f2bf_rne(float f) {
    union { float f; unsigned u; } v; v.f = f;
    unsigned u = v.u;
    u += 0x7FFFu + ((u >> 16) & 1u);   // round-to-nearest-even
    return (unsigned short)(u >> 16);
}

// CDNA5 async global->LDS copy (ASYNCcnt-tracked). %0 = per-lane LDS byte
// address (VGPR), %1 = 64-bit global address (VGPR pair).
__device__ __forceinline__ void async_copy_b128(unsigned lds, const unsigned short* g) {
    asm volatile("global_load_async_to_lds_b128 %0, %1, off"
                 :: "v"(lds), "v"(g) : "memory");
}
__device__ __forceinline__ void wait_async_le6() {
    asm volatile("s_wait_asynccnt 0x6" ::: "memory");
}
__device__ __forceinline__ void wait_async_0() {
    asm volatile("s_wait_asynccnt 0x0" ::: "memory");
}

// ---------------------------------------------------------------------------
// Kernel 1: L2-normalize rows of audio & label features, store as bf16.
// ---------------------------------------------------------------------------
__global__ __launch_bounds__(256) void k_normalize(const float* __restrict__ af,
                                                   const float* __restrict__ lf,
                                                   unsigned short* __restrict__ a_bf,
                                                   unsigned short* __restrict__ l_bf)
{
    const int row = blockIdx.x;
    const int tid = threadIdx.x;
    const float* ap = af + (size_t)row * D_K;
    const float* lp = lf + (size_t)row * D_K;
    float a0 = ap[tid], a1 = ap[tid + 256];
    float l0 = lp[tid], l1 = lp[tid + 256];
    float sa = a0 * a0 + a1 * a1;
    float sl = l0 * l0 + l1 * l1;
    #pragma unroll
    for (int m = 1; m < 32; m <<= 1) {
        sa += __shfl_xor(sa, m, 32);
        sl += __shfl_xor(sl, m, 32);
    }
    __shared__ float red[2][8];
    __shared__ float inv[2];
    const int wave = tid >> 5, lane = tid & 31;
    if (lane == 0) { red[0][wave] = sa; red[1][wave] = sl; }
    __syncthreads();
    if (wave == 0) {
        float xa = (lane < 8) ? red[0][lane] : 0.f;
        float xl = (lane < 8) ? red[1][lane] : 0.f;
        #pragma unroll
        for (int m = 1; m < 8; m <<= 1) {
            xa += __shfl_xor(xa, m, 32);
            xl += __shfl_xor(xl, m, 32);
        }
        if (lane == 0) {
            inv[0] = 1.0f / fmaxf(sqrtf(xa), 1e-12f);
            inv[1] = 1.0f / fmaxf(sqrtf(xl), 1e-12f);
        }
    }
    __syncthreads();
    const float ia = inv[0], il = inv[1];
    unsigned short* aw = a_bf + (size_t)row * D_K;
    unsigned short* lw = l_bf + (size_t)row * D_K;
    aw[tid]       = f2bf_rne(a0 * ia);
    aw[tid + 256] = f2bf_rne(a1 * ia);
    lw[tid]       = f2bf_rne(l0 * il);
    lw[tid + 256] = f2bf_rne(l1 * il);
}

// ---------------------------------------------------------------------------
// Kernel 2: zero the four global accumulator arrays.
// ---------------------------------------------------------------------------
__global__ void k_zero(float* __restrict__ p, int n)
{
    int i = blockIdx.x * blockDim.x + threadIdx.x;
    if (i < n) p[i] = 0.f;
}

// ---------------------------------------------------------------------------
// Kernel 3: fused  S = (L @ A^T)/T -> exp -> masked row/col reductions.
// Block tile 128(M) x 256(N), 8 waves, wave tile 64x64 = 4x4 WMMA tiles.
// Operands staged into LDS once per block via double-buffered
// global_load_async_to_lds_b128 (ASYNCcnt), fragments read from LDS.
// ---------------------------------------------------------------------------
__global__ __launch_bounds__(256) void k_gemm_loss(
    const unsigned short* __restrict__ l_bf,
    const unsigned short* __restrict__ a_bf,
    const int* __restrict__ labels,
    float* __restrict__ rowSum, float* __restrict__ rowPos,
    float* __restrict__ colSum, float* __restrict__ colPos)
{
    // 48 KB staging + reduction scratch (WGP has 320 KB LDS)
    __shared__ __align__(16) unsigned short bufA[2][128 * 32]; // 2 x 8 KB
    __shared__ __align__(16) unsigned short bufB[2][256 * 32]; // 2 x 16 KB
    __shared__ int   rowLab[128], colLab[256];
    __shared__ float rS[128], rP[128], cS[256], cP[256];

    const int tid = threadIdx.x;
    const int i0 = blockIdx.y * 128;   // row band  (label features)
    const int j0 = blockIdx.x * 256;   // col band  (audio features)

    if (tid < 128) { rowLab[tid] = labels[i0 + tid]; rS[tid] = 0.f; rP[tid] = 0.f; }
    colLab[tid] = labels[j0 + tid];  cS[tid] = 0.f;  cP[tid] = 0.f;

    const unsigned short* gA = l_bf + (size_t)i0 * D_K;
    const unsigned short* gB = a_bf + (size_t)j0 * D_K;
    const unsigned ldsA0 = (unsigned)(uintptr_t)&bufA[0][0];
    const unsigned ldsB0 = (unsigned)(uintptr_t)&bufB[0][0];

    const int wave = tid >> 5, lane = tid & 31;
    const int wm = wave >> 2;          // 2 row bands of 64
    const int wn = wave & 3;           // 4 col bands of 64
    const int half = lane >> 4, r = lane & 15;

    // ---- stage k-step `ks` into buffer `b` (6 async b128 per thread) ----
    auto stage = [&](int b, int k0) {
        const unsigned la = ldsA0 + (unsigned)b * (128 * 32 * 2);
        const unsigned lb = ldsB0 + (unsigned)b * (256 * 32 * 2);
        #pragma unroll
        for (int j = 0; j < 2; ++j) {               // A: 128 rows x 64 B
            int ch = tid + j * 256;
            int row = ch >> 2, o = ch & 3;
            async_copy_b128(la + ch * 16, gA + (size_t)row * D_K + k0 + o * 8);
        }
        #pragma unroll
        for (int j = 0; j < 4; ++j) {               // B: 256 rows x 64 B
            int ch = tid + j * 256;
            int row = ch >> 2, o = ch & 3;
            async_copy_b128(lb + ch * 16, gB + (size_t)row * D_K + k0 + o * 8);
        }
    };

    v8f acc[4][4] = {};

    stage(0, 0);
    for (int ks = 0; ks < NK; ++ks) {
        const int b = ks & 1;
        if (ks + 1 < NK) { stage(b ^ 1, (ks + 1) * 32); wait_async_le6(); }
        else             { wait_async_0(); }
        __syncthreads();                 // stage `ks` visible to all waves

        const unsigned short* sA = &bufA[b][0];
        const unsigned short* sB = &bufB[b][0];

        v16bf aF[4], bF[4];
        #pragma unroll
        for (int mt = 0; mt < 4; ++mt) {
            const unsigned short* p = sA + (wm * 64 + mt * 16 + r) * 32 + half * 8;
            v8bf lo = *(const v8bf*)(p);
            v8bf hi = *(const v8bf*)(p + 16);
            aF[mt] = __builtin_shufflevector(lo, hi,
                        0, 1, 2, 3, 4, 5, 6, 7, 8, 9, 10, 11, 12, 13, 14, 15);
        }
        #pragma unroll
        for (int nt = 0; nt < 4; ++nt) {
            const unsigned short* p = sB + (wn * 64 + nt * 16 + r) * 32 + half * 8;
            v8bf lo = *(const v8bf*)(p);
            v8bf hi = *(const v8bf*)(p + 16);
            bF[nt] = __builtin_shufflevector(lo, hi,
                        0, 1, 2, 3, 4, 5, 6, 7, 8, 9, 10, 11, 12, 13, 14, 15);
        }
        #pragma unroll
        for (int mt = 0; mt < 4; ++mt)
            #pragma unroll
            for (int nt = 0; nt < 4; ++nt)
                acc[mt][nt] = __builtin_amdgcn_wmma_f32_16x16x32_bf16(
                    false, aF[mt], false, bF[nt],
                    (short)0, acc[mt][nt], false, false);

        __syncthreads();                 // all reads done before next overwrite
    }

    // ---- epilogue: exp(S/T), positive mask, row/col partial sums ----
    int labC[4];
    #pragma unroll
    for (int nt = 0; nt < 4; ++nt)
        labC[nt] = colLab[wn * 64 + nt * 16 + r];

    float cs[4] = {}, cp[4] = {};
    #pragma unroll
    for (int mt = 0; mt < 4; ++mt) {
        int labRm[8];
        #pragma unroll
        for (int v = 0; v < 8; ++v)
            labRm[v] = rowLab[wm * 64 + mt * 16 + half * 8 + v];

        float rs[8] = {}, rp[8] = {};
        #pragma unroll
        for (int nt = 0; nt < 4; ++nt)
            #pragma unroll
            for (int v = 0; v < 8; ++v) {
                float e  = __expf(acc[mt][nt][v] * TEMP_INV);
                float pe = (labRm[v] == labC[nt]) ? e : 0.f;
                rs[v] += e;  rp[v] += pe;
                cs[nt] += e; cp[nt] += pe;
            }

        // row sums: butterfly over 16 lanes (stays within each half)
        #pragma unroll
        for (int v = 0; v < 8; ++v) {
            float x = rs[v], y = rp[v];
            #pragma unroll
            for (int m = 1; m < 16; m <<= 1) {
                x += __shfl_xor(x, m, 32);
                y += __shfl_xor(y, m, 32);
            }
            if (r == 0) {   // lanes 0 (M=v) and 16 (M=v+8)
                atomicAdd(&rS[wm * 64 + mt * 16 + half * 8 + v], x);
                atomicAdd(&rP[wm * 64 + mt * 16 + half * 8 + v], y);
            }
        }
    }

    // col sums: lane already summed its 8 rows per tile; merge halves
    #pragma unroll
    for (int nt = 0; nt < 4; ++nt) {
        float x = cs[nt] + __shfl_down(cs[nt], 16, 32);
        float y = cp[nt] + __shfl_down(cp[nt], 16, 32);
        if (half == 0) {
            atomicAdd(&cS[wn * 64 + nt * 16 + r], x);
            atomicAdd(&cP[wn * 64 + nt * 16 + r], y);
        }
    }

    __syncthreads();
    // one global f32 atomic per row/col per workgroup
    if (tid < 128) {
        atomicAdd(&rowSum[i0 + tid], rS[tid]);
        atomicAdd(&rowPos[i0 + tid], rP[tid]);
    }
    atomicAdd(&colSum[j0 + tid], cS[tid]);
    atomicAdd(&colPos[j0 + tid], cP[tid]);
}

// ---------------------------------------------------------------------------
// Kernel 4: final loss = mean_i -( log(rowPos/(rowSum+eps)+eps)
//                                + log(colPos/(colSum+eps)+eps) )
// ---------------------------------------------------------------------------
__global__ __launch_bounds__(256) void k_finalize(const float* __restrict__ rowSum,
                                                  const float* __restrict__ rowPos,
                                                  const float* __restrict__ colSum,
                                                  const float* __restrict__ colPos,
                                                  float* __restrict__ out)
{
    const int tid = threadIdx.x;
    float local = 0.f;
    for (int i = tid; i < B_N; i += 256) {
        float pl = rowPos[i] / (rowSum[i] + EPS_F);
        float pa = colPos[i] / (colSum[i] + EPS_F);
        local += -(__logf(pl + EPS_F) + __logf(pa + EPS_F));
    }
    #pragma unroll
    for (int m = 1; m < 32; m <<= 1) local += __shfl_xor(local, m, 32);
    __shared__ float red[8];
    const int wave = tid >> 5, lane = tid & 31;
    if (lane == 0) red[wave] = local;
    __syncthreads();
    if (tid == 0) {
        float s = 0.f;
        #pragma unroll
        for (int w = 0; w < 8; ++w) s += red[w];
        out[0] = s / (float)B_N;
    }
}

// ---------------------------------------------------------------------------
extern "C" void kernel_launch(void* const* d_in, const int* in_sizes, int n_in,
                              void* d_out, int out_size, void* d_ws, size_t ws_size,
                              hipStream_t stream)
{
    const float* audio = (const float*)d_in[0];
    const float* label = (const float*)d_in[1];
    const int*   cls   = (const int*)d_in[2];

    char* ws = (char*)d_ws;
    unsigned short* l_bf = (unsigned short*)ws;                           // 8 MiB
    unsigned short* a_bf = (unsigned short*)(ws + (size_t)B_N * D_K * 2); // 8 MiB
    float* rowSum = (float*)(ws + (size_t)B_N * D_K * 4);
    float* rowPos = rowSum + B_N;
    float* colSum = rowPos + B_N;
    float* colPos = colSum + B_N;

    k_normalize<<<B_N, 256, 0, stream>>>(audio, label, a_bf, l_bf);
    k_zero<<<(4 * B_N + 255) / 256, 256, 0, stream>>>(rowSum, 4 * B_N);
    dim3 grid(B_N / 256, B_N / 128);   // x: col blocks (256), y: row blocks (128)
    k_gemm_loss<<<grid, 256, 0, stream>>>(l_bf, a_bf, cls,
                                          rowSum, rowPos, colSum, colPos);
    k_finalize<<<1, 256, 0, stream>>>(rowSum, rowPos, colSum, colPos, (float*)d_out);
}
